// HeteroRGCN_28432683499963
// MI455X (gfx1250) — matrix operand order repaired and checked by reading, not compiled, not run on hardware
//
#include <hip/hip_runtime.h>
#include <hip/hip_bf16.h>

// ---------------------------------------------------------------------------
// HeteroRGCN forward, pruned to the live dataflow:
//   layer1: Wh = x_target @ w1[{1,3}] + b1  -> scatter-mean -> h1_card / h1_ip
//   layer2: Wh = lrelu(h1_{card,ip}) @ w2[{0,2}] + b2 -> scatter-mean -> h2_t
//   out   : softmax(h2_t @ w_out + b_out)
// Dense GEMMs use v_wmma_f32_16x16x32_bf16 (wave32, 16x16 tiles), W staged in
// LDS in B-operand layout. Irregular part uses global_atomic_add_f32.
// ---------------------------------------------------------------------------

typedef float  v8f   __attribute__((ext_vector_type(8)));
typedef __bf16 v16bf __attribute__((ext_vector_type(16)));
typedef unsigned short bf16x16 __attribute__((ext_vector_type(16)));

__device__ __forceinline__ unsigned short f2bf(float f) {
    // round-to-nearest-even f32 -> bf16
    unsigned int u = __builtin_bit_cast(unsigned int, f);
    unsigned int r = u + 0x7fffu + ((u >> 16) & 1u);
    return (unsigned short)(r >> 16);
}

__device__ __forceinline__ float lrelu(float x) {
    return x > 0.0f ? x : 0.01f * x;
}

// ---------------------------------------------------------------------------
// GEMM: Y[M,128] = op(X[M,128]) @ W[128,128] + bias[128], op = id | leaky_relu
// Block = 256 threads = 8 waves; each wave computes a 16x128 strip.
// ---------------------------------------------------------------------------
__global__ __launch_bounds__(256) void rgcn_gemm_wmma(
    const float* __restrict__ X, const float* __restrict__ W,
    const float* __restrict__ bias, float* __restrict__ Y,
    int M, int leaky)
{
    // W packed as bf16 in B-operand order: tile t=(kk*8+nt), lane L, elem e:
    //   sW[(t*32+L)*16+e] = W[kk*32 + 16*(L>>4) + e][nt*16 + (L&15)]
    __shared__ __align__(32) unsigned short sW[128 * 128];

    const int tid = threadIdx.x;
    for (int p = tid; p < 128 * 128; p += 256) {
        int e  = p & 15;
        int L  = (p >> 4) & 31;
        int t  = p >> 9;            // 0..31
        int kk = t >> 3, nt = t & 7;
        int K  = kk * 32 + ((L >> 4) << 4) + e;
        int N  = nt * 16 + (L & 15);
        sW[p] = f2bf(W[K * 128 + N]);
    }
    __syncthreads();

    const int wave = tid >> 5;
    const int lane = tid & 31;
    const int half = lane >> 4;
    const int m    = lane & 15;
    const long rowBase = (long)blockIdx.x * 128 + wave * 16;
    const long rowA    = rowBase + m;

    v8f acc[8];
#pragma unroll
    for (int nt = 0; nt < 8; ++nt)
#pragma unroll
        for (int j = 0; j < 8; ++j) acc[nt][j] = 0.0f;

#pragma unroll
    for (int kk = 0; kk < 4; ++kk) {
        union { v16bf v; bf16x16 u; } A;
        if (rowA < M) {
            // A fragment: K=16*half+e
            const float* xr = X + rowA * 128 + kk * 32;
            float4 f0 = *(const float4*)(xr + 16 * half);
            float4 f1 = *(const float4*)(xr + 16 * half + 4);
            float4 f2 = *(const float4*)(xr + 16 * half + 8);
            float4 f3 = *(const float4*)(xr + 16 * half + 12);
            if (leaky) {
                f0.x = lrelu(f0.x); f0.y = lrelu(f0.y); f0.z = lrelu(f0.z); f0.w = lrelu(f0.w);
                f1.x = lrelu(f1.x); f1.y = lrelu(f1.y); f1.z = lrelu(f1.z); f1.w = lrelu(f1.w);
                f2.x = lrelu(f2.x); f2.y = lrelu(f2.y); f2.z = lrelu(f2.z); f2.w = lrelu(f2.w);
                f3.x = lrelu(f3.x); f3.y = lrelu(f3.y); f3.z = lrelu(f3.z); f3.w = lrelu(f3.w);
            }
            A.u[0] = f2bf(f0.x); A.u[1] = f2bf(f0.y); A.u[2]  = f2bf(f0.z); A.u[3]  = f2bf(f0.w);
            A.u[4] = f2bf(f1.x); A.u[5] = f2bf(f1.y); A.u[6]  = f2bf(f1.z); A.u[7]  = f2bf(f1.w);
            A.u[8] = f2bf(f2.x); A.u[9] = f2bf(f2.y); A.u[10] = f2bf(f2.z); A.u[11] = f2bf(f2.w);
            A.u[12] = f2bf(f3.x); A.u[13] = f2bf(f3.y); A.u[14] = f2bf(f3.z); A.u[15] = f2bf(f3.w);
        } else {
#pragma unroll
            for (int e = 0; e < 16; ++e) A.u[e] = 0;
        }
#pragma unroll
        for (int nt = 0; nt < 8; ++nt) {
            const v16bf b =
                *reinterpret_cast<const v16bf*>(&sW[(((kk << 3) + nt) * 32 + lane) << 4]);
            acc[nt] = __builtin_amdgcn_wmma_f32_16x16x32_bf16(
                /*neg_a=*/false, A.v, /*neg_b=*/false, b,
                /*c_mod=*/(short)0, acc[nt], /*reuse_a=*/false, /*reuse_b=*/false);
        }
    }

    // C/D layout: vgpr j -> row = rowBase + j + 8*half ; col = nt*16 + (lane&15)
#pragma unroll
    for (int nt = 0; nt < 8; ++nt) {
        const int col = nt * 16 + m;
        const float bv = bias[col];
#pragma unroll
        for (int j = 0; j < 8; ++j) {
            long r = rowBase + j + 8 * half;
            if (r < M) Y[r * 128 + col] = acc[nt][j] + bv;
        }
    }
}

// ---------------------------------------------------------------------------
// Zero-fill
// ---------------------------------------------------------------------------
__global__ void rgcn_fill(float* __restrict__ p, long n) {
    long i = (long)blockIdx.x * blockDim.x + threadIdx.x;
    if (i < n) p[i] = 0.0f;
}

// ---------------------------------------------------------------------------
// Scatter: sum[dst] += Wh[src] (128 feats), deg[dst] += 1. One wave per edge.
// ---------------------------------------------------------------------------
__global__ __launch_bounds__(256) void rgcn_scatter(
    const float* __restrict__ Wh, const int* __restrict__ src,
    const int* __restrict__ dst, float* __restrict__ sum,
    float* __restrict__ deg, int E)
{
    int slot = threadIdx.x >> 5;
    int lane = threadIdx.x & 31;
    int e = blockIdx.x * 8 + slot;
    if (e >= E) return;
    int s = src[e], d = dst[e];
    float4 v = ((const float4*)(Wh + (long)s * 128))[lane];
    float* pd = sum + (long)d * 128 + lane * 4;
    atomicAdd(pd + 0, v.x);
    atomicAdd(pd + 1, v.y);
    atomicAdd(pd + 2, v.z);
    atomicAdd(pd + 3, v.w);
    if (lane == 0) atomicAdd(deg + d, 1.0f);
}

// ---------------------------------------------------------------------------
// out[i] (=|+=) sum[i] / max(deg[row],1)
// ---------------------------------------------------------------------------
__global__ void rgcn_combine(const float* __restrict__ sum,
                             const float* __restrict__ deg,
                             float* __restrict__ out, long n, int accumulate)
{
    long i = (long)blockIdx.x * blockDim.x + threadIdx.x;
    if (i >= n) return;
    float mval = sum[i] / fmaxf(deg[i >> 7], 1.0f);
    out[i] = accumulate ? (out[i] + mval) : mval;
}

// ---------------------------------------------------------------------------
// logits = h @ w_out[128,2] + b_out ; softmax over 2
// ---------------------------------------------------------------------------
__global__ void rgcn_out_softmax(const float* __restrict__ h,
                                 const float* __restrict__ w_out,
                                 const float* __restrict__ b_out,
                                 float* __restrict__ out, int N)
{
    int row = blockIdx.x * blockDim.x + threadIdx.x;
    if (row >= N) return;
    const float4* hr = (const float4*)(h + (long)row * 128);
    float a0 = b_out[0], a1 = b_out[1];
#pragma unroll 8
    for (int j = 0; j < 32; ++j) {
        float4 v = hr[j];
        int k = j * 4;
        a0 += v.x * w_out[(k + 0) * 2] + v.y * w_out[(k + 1) * 2] +
              v.z * w_out[(k + 2) * 2] + v.w * w_out[(k + 3) * 2];
        a1 += v.x * w_out[(k + 0) * 2 + 1] + v.y * w_out[(k + 1) * 2 + 1] +
              v.z * w_out[(k + 2) * 2 + 1] + v.w * w_out[(k + 3) * 2 + 1];
    }
    float mx = fmaxf(a0, a1);
    float e0 = __expf(a0 - mx), e1 = __expf(a1 - mx);
    float inv = 1.0f / (e0 + e1);
    out[(long)row * 2 + 0] = e0 * inv;
    out[(long)row * 2 + 1] = e1 * inv;
}

// ---------------------------------------------------------------------------
extern "C" void kernel_launch(void* const* d_in, const int* in_sizes, int n_in,
                              void* d_out, int out_size, void* d_ws, size_t ws_size,
                              hipStream_t stream) {
    const float* x_target = (const float*)d_in[0];
    // d_in[1] (x_card) and d_in[2] (x_ip) are dead: they only feed h1["target"],
    // which nothing downstream of the returned output consumes.
    const float* w1    = (const float*)d_in[3];   // [4,128,128]
    const float* b1    = (const float*)d_in[4];   // [4,128]
    const float* w2    = (const float*)d_in[5];   // [4,128,128]
    const float* b2    = (const float*)d_in[6];   // [4,128]
    const float* w_out = (const float*)d_in[7];   // [128,2]
    const float* b_out = (const float*)d_in[8];   // [2]
    const int* src0 = (const int*)d_in[9];
    const int* dst0 = (const int*)d_in[10];
    const int* src1 = (const int*)d_in[11];
    const int* dst1 = (const int*)d_in[12];
    const int* src2 = (const int*)d_in[13];
    const int* dst2 = (const int*)d_in[14];
    const int* src3 = (const int*)d_in[15];
    const int* dst3 = (const int*)d_in[16];

    const int N_T = in_sizes[0] / 128;
    const int N_C = in_sizes[1] / 128;
    const int N_I = in_sizes[2] / 128;
    const int E   = in_sizes[9];

    // workspace layout (floats)
    float* ws  = (float*)d_ws;
    float* Wh  = ws;                          // N_T*128 (max GEMM output)
    float* sum = Wh  + (long)N_T * 128;       // N_T*128 (reused per relation)
    float* deg = sum + (long)N_T * 128;       // N_T
    float* h1c = deg + N_T;                   // N_C*128
    float* h1i = h1c + (long)N_C * 128;       // N_I*128
    float* h2t = h1i + (long)N_I * 128;       // N_T*128
    (void)ws_size; (void)n_in; (void)out_size;

    const int TB = 256;
    const int gE = (E + 7) / 8;
    auto gemmGrid = [](int M) { return (M + 127) / 128; };
    auto fillGrid = [](long n) { return (int)((n + 255) / 256); };

    // ---- layer 1, etype1: target -> card ---------------------------------
    rgcn_gemm_wmma<<<gemmGrid(N_T), TB, 0, stream>>>(x_target, w1 + 1 * 16384,
                                                     b1 + 1 * 128, Wh, N_T, 0);
    rgcn_fill<<<fillGrid((long)N_C * 128), TB, 0, stream>>>(sum, (long)N_C * 128);
    rgcn_fill<<<fillGrid(N_C), TB, 0, stream>>>(deg, N_C);
    rgcn_scatter<<<gE, TB, 0, stream>>>(Wh, src1, dst1, sum, deg, E);
    rgcn_combine<<<fillGrid((long)N_C * 128), TB, 0, stream>>>(sum, deg, h1c,
                                                               (long)N_C * 128, 0);

    // ---- layer 1, etype3: target -> ip -----------------------------------
    rgcn_gemm_wmma<<<gemmGrid(N_T), TB, 0, stream>>>(x_target, w1 + 3 * 16384,
                                                     b1 + 3 * 128, Wh, N_T, 0);
    rgcn_fill<<<fillGrid((long)N_I * 128), TB, 0, stream>>>(sum, (long)N_I * 128);
    rgcn_fill<<<fillGrid(N_I), TB, 0, stream>>>(deg, N_I);
    rgcn_scatter<<<gE, TB, 0, stream>>>(Wh, src3, dst3, sum, deg, E);
    rgcn_combine<<<fillGrid((long)N_I * 128), TB, 0, stream>>>(sum, deg, h1i,
                                                               (long)N_I * 128, 0);

    // ---- layer 2, etype0: card -> target (leaky fused on load) -----------
    rgcn_gemm_wmma<<<gemmGrid(N_C), TB, 0, stream>>>(h1c, w2 + 0 * 16384,
                                                     b2 + 0 * 128, Wh, N_C, 1);
    rgcn_fill<<<fillGrid((long)N_T * 128), TB, 0, stream>>>(sum, (long)N_T * 128);
    rgcn_fill<<<fillGrid(N_T), TB, 0, stream>>>(deg, N_T);
    rgcn_scatter<<<gE, TB, 0, stream>>>(Wh, src0, dst0, sum, deg, E);
    rgcn_combine<<<fillGrid((long)N_T * 128), TB, 0, stream>>>(sum, deg, h2t,
                                                               (long)N_T * 128, 0);

    // ---- layer 2, etype2: ip -> target -----------------------------------
    rgcn_gemm_wmma<<<gemmGrid(N_I), TB, 0, stream>>>(h1i, w2 + 2 * 16384,
                                                     b2 + 2 * 128, Wh, N_I, 1);
    rgcn_fill<<<fillGrid((long)N_T * 128), TB, 0, stream>>>(sum, (long)N_T * 128);
    rgcn_fill<<<fillGrid(N_T), TB, 0, stream>>>(deg, N_T);
    rgcn_scatter<<<gE, TB, 0, stream>>>(Wh, src2, dst2, sum, deg, E);
    rgcn_combine<<<fillGrid((long)N_T * 128), TB, 0, stream>>>(sum, deg, h2t,
                                                               (long)N_T * 128, 1);

    // ---- output head ------------------------------------------------------
    rgcn_out_softmax<<<(N_T + TB - 1) / TB, TB, 0, stream>>>(h2t, w_out, b_out,
                                                             (float*)d_out, N_T);
}
